// GAM_70918499992174
// MI455X (gfx1250) — compile-verified, matrix operand order checked by reading, not run on hardware
//
#include <hip/hip_runtime.h>

#define IMG_H 128
#define IMG_W 256
#define HW_N  (IMG_H * IMG_W)   // 32768
#define BATCH 4
#define BN_EPS 1e-5f

typedef _Float16 h8  __attribute__((ext_vector_type(8)));
typedef _Float16 h16 __attribute__((ext_vector_type(16)));
typedef float    f8  __attribute__((ext_vector_type(8)));

// ---------------------------------------------------------------------------
// Position maps: p = w_p @ [loc_w; loc_h] + b_p  (and same for guidance pg)
// ---------------------------------------------------------------------------
__global__ void pos_kernel(const float* __restrict__ wp, const float* __restrict__ bp,
                           const float* __restrict__ wpg, const float* __restrict__ bpg,
                           float* __restrict__ p, float* __restrict__ pg) {
    int n = blockIdx.x * blockDim.x + threadIdx.x;
    if (n >= HW_N) return;
    int h = n / IMG_W, w = n % IMG_W;
    float lw = -1.0f + 2.0f * (float)w / (float)(IMG_W - 1);
    float lh = -1.0f + 2.0f * (float)h / (float)(IMG_H - 1);
    p[n]          = wp[0] * lw + wp[1] * lh + bp[0];
    p[HW_N + n]   = wp[2] * lw + wp[3] * lh + bp[1];
    pg[n]         = wpg[0] * lw + wpg[1] * lh + bpg[0];
    pg[HW_N + n]  = wpg[2] * lw + wpg[3] * lh + bpg[1];
}

// ---------------------------------------------------------------------------
// Pack [w_c1; w_c2; w_c3] -> Wx[80,64]+bx[80], [w_g1; w_g2] -> Wg[16,128]+bg[16]
// ---------------------------------------------------------------------------
__global__ void pack_weights(const float* __restrict__ wc1, const float* __restrict__ bc1,
                             const float* __restrict__ wc2, const float* __restrict__ bc2,
                             const float* __restrict__ wc3, const float* __restrict__ bc3,
                             const float* __restrict__ wg1, const float* __restrict__ bg1,
                             const float* __restrict__ wg2, const float* __restrict__ bg2,
                             float* __restrict__ Wx, float* __restrict__ bx,
                             float* __restrict__ Wg, float* __restrict__ bg) {
    int i = blockIdx.x * blockDim.x + threadIdx.x;
    if (i < 80 * 64) {
        int m = i / 64, k = i % 64;
        float v = (m < 8) ? wc1[m * 64 + k]
                : (m < 16) ? wc2[(m - 8) * 64 + k]
                : wc3[(m - 16) * 64 + k];
        Wx[i] = v;
    } else if (i < 80 * 64 + 80) {
        int m = i - 80 * 64;
        bx[m] = (m < 8) ? bc1[m] : (m < 16) ? bc2[m - 8] : bc3[m - 16];
    } else if (i < 80 * 64 + 80 + 16 * 128) {
        int j = i - (80 * 64 + 80);
        int m = j / 128, k = j % 128;
        Wg[j] = (m < 8) ? wg1[m * 128 + k] : wg2[(m - 8) * 128 + k];
    } else if (i < 80 * 64 + 80 + 16 * 128 + 16) {
        int m = i - (80 * 64 + 80 + 16 * 128);
        bg[m] = (m < 8) ? bg1[m] : bg2[m - 8];
    }
}

// ---------------------------------------------------------------------------
// 1x1 conv as WMMA GEMM:  Y[b, M, N] = W[M, KDIM] @ X[b, KDIM, N] + bias
// Block = 128 threads (4 wave32). Each wave owns one 16-column N tile and
// all MTILES 16-row M tiles. Weights + X tile staged in LDS as f16.
// All A/B fragments are preloaded into VGPRs before the WMMA chain so the
// matrix pipe is not gated by per-WMMA s_wait_dscnt.
// ---------------------------------------------------------------------------
template <int KDIM, int MTILES>
__global__ void __launch_bounds__(128)
conv1x1_wmma(const float* __restrict__ X, const float* __restrict__ Wm,
             const float* __restrict__ bias, float* __restrict__ Y) {
    constexpr int M      = MTILES * 16;
    constexpr int KP     = KDIM + 8;      // padded f16 row stride (multiple of 8)
    constexpr int KSTEPS = KDIM / 32;

    __shared__ __align__(16) _Float16 sW[M * KP];
    __shared__ __align__(16) _Float16 sX[4 * 16 * KP];

    const int tid  = threadIdx.x;
    const int lane = tid & 31;
    const int wave = tid >> 5;
    const int b    = blockIdx.y;
    const int c0   = blockIdx.x * 64 + wave * 16;   // first N column for this wave
    const int col  = lane & 15;                     // A row index / B column index
    const int half = lane >> 4;                     // 0: lanes 0-15, 1: lanes 16-31
    const int hk   = half * 8;                      // K sub-offset per ISA f16 layout

    // Stage weights (f32 -> f16), cooperative across the block.
    for (int i = tid; i < M * KDIM; i += 128) {
        int m = i / KDIM, k = i % KDIM;
        sW[m * KP + k] = (_Float16)Wm[i];
    }

    // Stage X tile transposed into LDS. Each lane owns a 4-column group and
    // loads float4 (16B) rows -> 4x fewer vmem instructions, same coalescing.
    const float* Xb = X + (size_t)b * KDIM * HW_N;
    _Float16* sXw = sX + wave * 16 * KP;
    const int cg = lane & 3;        // column group of 4
    const int r0 = lane >> 2;       // starting row (0..7)
    for (int k = r0; k < KDIM; k += 8) {
        const float4 xv = *(const float4*)&Xb[(size_t)k * HW_N + c0 + cg * 4];
        sXw[(cg * 4 + 0) * KP + k] = (_Float16)xv.x;
        sXw[(cg * 4 + 1) * KP + k] = (_Float16)xv.y;
        sXw[(cg * 4 + 2) * KP + k] = (_Float16)xv.z;
        sXw[(cg * 4 + 3) * KP + k] = (_Float16)xv.w;
    }
    __syncthreads();

    // Preload every fragment (fits comfortably in VGPRs: worst case
    // 5*2 A-frags (80) + 2 B-frags (16) + 5 accs (40) ~ 140 VGPRs).
    h16 bfrag[KSTEPS];
    h16 afrag[KSTEPS][MTILES];
#pragma unroll
    for (int ks = 0; ks < KSTEPS; ++ks) {
        const int kb = ks * 32;
        h8 blo = *(const h8*)&sXw[col * KP + kb + hk];
        h8 bhi = *(const h8*)&sXw[col * KP + kb + 16 + hk];
        bfrag[ks] = __builtin_shufflevector(blo, bhi, 0, 1, 2, 3, 4, 5, 6, 7,
                                                      8, 9, 10, 11, 12, 13, 14, 15);
#pragma unroll
        for (int t = 0; t < MTILES; ++t) {
            const _Float16* wrow = &sW[(t * 16 + col) * KP + kb + hk];
            h8 alo = *(const h8*)wrow;
            h8 ahi = *(const h8*)(wrow + 16);
            afrag[ks][t] = __builtin_shufflevector(alo, ahi, 0, 1, 2, 3, 4, 5, 6, 7,
                                                             8, 9, 10, 11, 12, 13, 14, 15);
        }
    }

    f8 acc[MTILES];
#pragma unroll
    for (int t = 0; t < MTILES; ++t) acc[t] = {};

    // Back-to-back WMMA chain (no LDS dependency inside the loop).
#pragma unroll
    for (int ks = 0; ks < KSTEPS; ++ks) {
#pragma unroll
        for (int t = 0; t < MTILES; ++t) {
            acc[t] = __builtin_amdgcn_wmma_f32_16x16x32_f16(
                false, afrag[ks][t], false, bfrag[ks], (short)0, acc[t], false, false);
        }
    }

    // C layout: VGPR r -> M = r + 8*half, lane&15 -> N. Coalesced stores.
    float* Yb = Y + (size_t)b * M * HW_N;
#pragma unroll
    for (int t = 0; t < MTILES; ++t) {
#pragma unroll
        for (int r = 0; r < 8; ++r) {
            int m = t * 16 + r + half * 8;
            Yb[(size_t)m * HW_N + c0 + col] = acc[t][r] + bias[m];
        }
    }
}

// ---------------------------------------------------------------------------
// Fused local-relation branch: for each pixel, 9-tap logits
// (sub + BN1 + ReLU + 8x10 matmul + BN2 + ReLU + 8->1), softmax over taps,
// then weighted 3x3 aggregation over 64 channels of v.
// ---------------------------------------------------------------------------
__global__ void __launch_bounds__(256)
branch_kernel(const float* __restrict__ q, size_t qStride,
              const float* __restrict__ kk, size_t kStride,
              const float* __restrict__ v, size_t vStride,
              const float* __restrict__ p,
              const float* __restrict__ bn1g, const float* __restrict__ bn1b,
              const float* __restrict__ bn1m, const float* __restrict__ bn1v,
              const float* __restrict__ wa,
              const float* __restrict__ bn2g, const float* __restrict__ bn2b,
              const float* __restrict__ bn2m, const float* __restrict__ bn2v,
              const float* __restrict__ wb, const float* __restrict__ bwb,
              float* __restrict__ out) {
    const int idx = blockIdx.x * blockDim.x + threadIdx.x;   // over BATCH*HW_N exactly
    const int b = idx / HW_N;
    const int n = idx % HW_N;
    const int h = n / IMG_W, w = n % IMG_W;

    // Fold BN into scale+shift (uniform -> scalar loads).
    float s1[10], t1[10];
#pragma unroll
    for (int c = 0; c < 10; ++c) {
        float s = bn1g[c] * rsqrtf(bn1v[c] + BN_EPS);
        s1[c] = s; t1[c] = bn1b[c] - bn1m[c] * s;
    }
    float s2[8], t2[8];
#pragma unroll
    for (int r = 0; r < 8; ++r) {
        float s = bn2g[r] * rsqrtf(bn2v[r] + BN_EPS);
        s2[r] = s; t2[r] = bn2b[r] - bn2m[r] * s;
    }

    float qv[8];
#pragma unroll
    for (int r = 0; r < 8; ++r) qv[r] = q[(size_t)b * qStride + (size_t)r * HW_N + n];
    const float p0 = p[n], p1 = p[HW_N + n];

    float wgt[9];
    int   nn[9];
    bool  ok[9];
#pragma unroll
    for (int t = 0; t < 9; ++t) {
        const int dy = t / 3 - 1, dx = t % 3 - 1;
        const int hh = h + dy, ww = w + dx;
        const bool in = (hh >= 0) & (hh < IMG_H) & (ww >= 0) & (ww < IMG_W);
        const int m = hh * IMG_W + ww;
        nn[t] = m; ok[t] = in;

        float f[10];
#pragma unroll
        for (int r = 0; r < 8; ++r)
            f[r] = qv[r] - (in ? kk[(size_t)b * kStride + (size_t)r * HW_N + m] : 0.0f);
        f[8] = p0 - (in ? p[m] : 0.0f);
        f[9] = p1 - (in ? p[HW_N + m] : 0.0f);

        float h1[10];
#pragma unroll
        for (int c = 0; c < 10; ++c) h1[c] = fmaxf(fmaf(f[c], s1[c], t1[c]), 0.0f);

        float lg = bwb[0];
#pragma unroll
        for (int r = 0; r < 8; ++r) {
            float a = 0.0f;
#pragma unroll
            for (int c = 0; c < 10; ++c) a = fmaf(wa[r * 10 + c], h1[c], a);
            a = fmaxf(fmaf(a, s2[r], t2[r]), 0.0f);
            lg = fmaf(wb[r], a, lg);
        }
        wgt[t] = lg;
    }

    // softmax over the 9 taps (OOB taps participate, matching zero-padded unfold)
    float mx = wgt[0];
#pragma unroll
    for (int t = 1; t < 9; ++t) mx = fmaxf(mx, wgt[t]);
    float sum = 0.0f;
#pragma unroll
    for (int t = 0; t < 9; ++t) { wgt[t] = __expf(wgt[t] - mx); sum += wgt[t]; }
    const float inv = 1.0f / sum;
#pragma unroll
    for (int t = 0; t < 9; ++t) wgt[t] *= inv;

    // weighted 3x3 aggregation of v (neighbor reads stay hot in L2)
    const float* vb = v + (size_t)b * vStride;
    float* ob = out + (size_t)b * 64 * HW_N;
    for (int c = 0; c < 64; ++c) {
        float a = 0.0f;
#pragma unroll
        for (int t = 0; t < 9; ++t)
            if (ok[t]) a = fmaf(wgt[t], vb[(size_t)c * HW_N + nn[t]], a);
        ob[(size_t)c * HW_N + n] = a;
    }
}

// ---------------------------------------------------------------------------
extern "C" void kernel_launch(void* const* d_in, const int* in_sizes, int n_in,
                              void* d_out, int out_size, void* d_ws, size_t ws_size,
                              hipStream_t stream) {
    const float* x    = (const float*)d_in[0];
    const float* g    = (const float*)d_in[1];
    const float* w_c1 = (const float*)d_in[2];  const float* b_c1 = (const float*)d_in[3];
    const float* w_c2 = (const float*)d_in[4];  const float* b_c2 = (const float*)d_in[5];
    const float* w_c3 = (const float*)d_in[6];  const float* b_c3 = (const float*)d_in[7];
    const float* w_p  = (const float*)d_in[8];  const float* b_p  = (const float*)d_in[9];
    const float* bn1g = (const float*)d_in[10]; const float* bn1b = (const float*)d_in[11];
    const float* bn1m = (const float*)d_in[12]; const float* bn1v = (const float*)d_in[13];
    const float* w_wa = (const float*)d_in[14];
    const float* bn2g = (const float*)d_in[15]; const float* bn2b = (const float*)d_in[16];
    const float* bn2m = (const float*)d_in[17]; const float* bn2v = (const float*)d_in[18];
    const float* w_wb = (const float*)d_in[19]; const float* b_wb = (const float*)d_in[20];
    const float* w_g1 = (const float*)d_in[21]; const float* b_g1 = (const float*)d_in[22];
    const float* w_g2 = (const float*)d_in[23]; const float* b_g2 = (const float*)d_in[24];
    const float* w_pg = (const float*)d_in[25]; const float* b_pg = (const float*)d_in[26];
    const float* gbn1g = (const float*)d_in[27]; const float* gbn1b = (const float*)d_in[28];
    const float* gbn1m = (const float*)d_in[29]; const float* gbn1v = (const float*)d_in[30];
    const float* w_gwa = (const float*)d_in[31];
    const float* gbn2g = (const float*)d_in[32]; const float* gbn2b = (const float*)d_in[33];
    const float* gbn2m = (const float*)d_in[34]; const float* gbn2v = (const float*)d_in[35];
    const float* w_gwb = (const float*)d_in[36]; const float* b_gwb = (const float*)d_in[37];

    float* ws = (float*)d_ws;
    size_t off = 0;
    float* p    = ws + off; off += 2 * HW_N;
    float* pg   = ws + off; off += 2 * HW_N;
    float* Wx   = ws + off; off += 80 * 64;
    float* bx   = ws + off; off += 80;
    float* Wg   = ws + off; off += 16 * 128;
    float* bg   = ws + off; off += 16 + 32;                     // small pad
    float* Y1   = ws + off; off += (size_t)BATCH * 80 * HW_N;   // q|k|v for branch 1
    float* Yg   = ws + off; off += (size_t)BATCH * 16 * HW_N;   // q|k for branch 2
    float* out1 = ws + off; off += (size_t)BATCH * 64 * HW_N;   // branch-1 output (v for branch 2)

    pos_kernel<<<HW_N / 256, 256, 0, stream>>>(w_p, b_p, w_pg, b_pg, p, pg);
    pack_weights<<<(80 * 64 + 80 + 16 * 128 + 16 + 255) / 256, 256, 0, stream>>>(
        w_c1, b_c1, w_c2, b_c2, w_c3, b_c3, w_g1, b_g1, w_g2, b_g2, Wx, bx, Wg, bg);

    // Projections via WMMA GEMM
    conv1x1_wmma<64, 5><<<dim3(HW_N / 64, BATCH), 128, 0, stream>>>(x, Wx, bx, Y1);
    conv1x1_wmma<128, 1><<<dim3(HW_N / 64, BATCH), 128, 0, stream>>>(g, Wg, bg, Yg);

    // Branch 1: q = Y1[0:8], k = Y1[8:16], v = Y1[16:80]
    branch_kernel<<<(size_t)BATCH * HW_N / 256, 256, 0, stream>>>(
        Y1, (size_t)80 * HW_N,
        Y1 + (size_t)8 * HW_N, (size_t)80 * HW_N,
        Y1 + (size_t)16 * HW_N, (size_t)80 * HW_N,
        p, bn1g, bn1b, bn1m, bn1v, w_wa, bn2g, bn2b, bn2m, bn2v, w_wb, b_wb, out1);

    // Branch 2 (guidance): q = Yg[0:8], k = Yg[8:16], v = out1
    branch_kernel<<<(size_t)BATCH * HW_N / 256, 256, 0, stream>>>(
        Yg, (size_t)16 * HW_N,
        Yg + (size_t)8 * HW_N, (size_t)16 * HW_N,
        out1, (size_t)64 * HW_N,
        pg, gbn1g, gbn1b, gbn1m, gbn1v, w_gwa, gbn2g, gbn2b, gbn2m, gbn2v,
        w_gwb, b_gwb, (float*)d_out);
}
